// AttentionFusionHead_21328807592542
// MI455X (gfx1250) — matrix operand order repaired and checked by reading, not compiled, not run on hardware
//
#include <hip/hip_runtime.h>
#include <hip/hip_bf16.h>

// ---------------------------------------------------------------------------
// AttentionFusionHead for MI455X (gfx1250): bf16 WMMA GEMM pipeline.
//   - All large GEMMs via v_wmma_f32_16x16x32_bf16 (f32 accumulate).
//   - Tile staging via GLOBAL_LOAD_ASYNC_TO_LDS_B128 + double-buffered LDS
//     (ASYNCcnt-tracked, VGPR-bypassing) when the builtin is available;
//     otherwise register-staged software-pipelined copies.
//   - Tiny S=2 attention + LayerNorm + residual/mean in f32 VALU kernels.
// ---------------------------------------------------------------------------

typedef __bf16 bf16_t;
typedef __attribute__((ext_vector_type(16))) __bf16 v16bf;
typedef __attribute__((ext_vector_type(8)))  float  v8f;
typedef int v4i __attribute__((vector_size(16)));   // matches builtin's V4i

#define AS1 __attribute__((address_space(1)))
#define AS3 __attribute__((address_space(3)))

#if defined(__gfx1250__) && __has_builtin(__builtin_amdgcn_global_load_async_to_lds_b128)
#define USE_ASYNC_LDS 1
#else
#define USE_ASYNC_LDS 0
#endif

__device__ __forceinline__ void async_wait0() {
#if __has_builtin(__builtin_amdgcn_s_wait_asynccnt)
  __builtin_amdgcn_s_wait_asynccnt(0);
#else
  asm volatile("s_wait_asynccnt 0x0" ::: "memory");
#endif
}

#if USE_ASYNC_LDS
__device__ __forceinline__ void async_copy_b128(const void* g, void* l) {
  __builtin_amdgcn_global_load_async_to_lds_b128(
      (AS1 v4i*)(void*)g, (AS3 v4i*)l, 0, 0);
}
#endif

#define BDIM    8192
#define CDIM    2048
#define TDIM    2048
#define GDIM    1024
#define ODIM    1024
#define HEADS   16
#define HD      128      // head dim
#define ROWS    (BDIM*2) // 16384 stacked rows

// ---- bf16 helpers (bit-exact RNE convert, no reliance on __bf16 casts) ----
__device__ __forceinline__ bf16_t f2bf(float f) {
  unsigned int u = __float_as_uint(f);
  unsigned int r = (u + 0x7FFFu + ((u >> 16) & 1u)) >> 16;
  unsigned short s = (unsigned short)r;
  return __builtin_bit_cast(bf16_t, s);
}
__device__ __forceinline__ float bf2f(bf16_t h) {
  unsigned short s = __builtin_bit_cast(unsigned short, h);
  return __uint_as_float((unsigned int)s << 16);
}

// ---------------------------------------------------------------------------
// f32 -> bf16 conversion (grid-stride)
// ---------------------------------------------------------------------------
__global__ __launch_bounds__(256) void cvt_f32_bf16(const float* __restrict__ in,
                                                    bf16_t* __restrict__ out,
                                                    size_t n) {
  size_t i = (size_t)blockIdx.x * blockDim.x + threadIdx.x;
  size_t stride = (size_t)gridDim.x * blockDim.x;
  for (; i < n; i += stride) out[i] = f2bf(in[i]);
}

// ---------------------------------------------------------------------------
// bf16 WMMA GEMM:  Y[M,N] = A[M,K] @ W[N,K]^T + bias
//   A row-major (lda), W row-major (ldw) == B fragment layout natively.
//   Writes f32 (Yf) and/or bf16 (Ybf) with leading dim ldy.
// Workgroup: 256 threads = 8 waves; tile 128x128, BK=64.
// Each wave: 64x32 block = 4x2 WMMA accumulators.
// ---------------------------------------------------------------------------
#define BM 128
#define BN 128
#define BK 64
#define LDT 72   // BK + 8 bf16 pad (16B) -> conflict-spread rows, 16B aligned
#define TILE_ELEMS (BM * LDT)

__device__ __forceinline__ v16bf frag_a(const bf16_t* __restrict__ s,
                                        int m, int kbase, int lane) {
  // A 16x32 bf16 layout: lane L -> row m0+L%16; lanes>=16 shift K by 8 within
  // each 16-wide K group; VGPR j holds K pair (g*16 + khalf + 2*(j%4)).
  v16bf f;
  const int khalf = (lane >> 4) << 3;
  const bf16_t* row = s + m * LDT;
#pragma unroll
  for (int j = 0; j < 8; ++j) {
    int k = kbase + ((j >> 2) << 4) + khalf + ((j & 3) << 1);
    f[2 * j]     = row[k];
    f[2 * j + 1] = row[k + 1];
  }
  return f;
}

__device__ __forceinline__ v16bf frag_b(const bf16_t* __restrict__ s,
                                        int n, int kbase, int lane) {
  // B 32x16 bf16 layout: lane L -> col n0+L%16; lanes>=16 shift K by 8 within
  // each 16-wide K group; VGPR j holds K pair (g*16 + khalf + 2*(j%4)).
  v16bf f;
  const int khalf = (lane >> 4) << 3;
  const bf16_t* row = s + n * LDT;
#pragma unroll
  for (int j = 0; j < 8; ++j) {
    int k = kbase + ((j >> 2) << 4) + khalf + ((j & 3) << 1);
    f[2 * j]     = row[k];
    f[2 * j + 1] = row[k + 1];
  }
  return f;
}

__device__ __forceinline__ void do_wmma(const bf16_t* __restrict__ cA,
                                        const bf16_t* __restrict__ cB,
                                        v8f acc[4][2],
                                        int lane, int waveM, int waveN) {
#pragma unroll
  for (int kk = 0; kk < BK; kk += 32) {
    v16bf aF[4], bF[2];
#pragma unroll
    for (int i = 0; i < 4; ++i)
      aF[i] = frag_a(cA, waveM * 64 + i * 16 + (lane & 15), kk, lane);
#pragma unroll
    for (int j = 0; j < 2; ++j)
      bF[j] = frag_b(cB, waveN * 32 + j * 16 + (lane & 15), kk, lane);
#pragma unroll
    for (int i = 0; i < 4; ++i)
#pragma unroll
      for (int j = 0; j < 2; ++j)
        acc[i][j] = __builtin_amdgcn_wmma_f32_16x16x32_bf16(
            false, aF[i], false, bF[j], (short)0, acc[i][j], false, false);
  }
}

#if USE_ASYNC_LDS
// Stage both 128x64 bf16 tiles for K-step k0 into LDS buffers (async copy,
// ASYNCcnt tracked, no VGPR round-trip). 8 x b128 per thread (4 A + 4 B).
__device__ __forceinline__ void stage_async(
    const bf16_t* __restrict__ A, int lda, int mBlk,
    const bf16_t* __restrict__ W, int ldw, int nBlk,
    bf16_t* sA, bf16_t* sB, int k0, int tid) {
#pragma unroll
  for (int i = 0; i < 4; ++i) {
    int idx = tid + i * 256;
    int row = idx >> 3, ch = (idx & 7) << 3;
    async_copy_b128((const void*)(A + (size_t)(mBlk + row) * lda + k0 + ch),
                    (void*)(sA + row * LDT + ch));
    async_copy_b128((const void*)(W + (size_t)(nBlk + row) * ldw + k0 + ch),
                    (void*)(sB + row * LDT + ch));
  }
}
#endif

__global__ __launch_bounds__(256) void gemm_bf16_wmma(
    const bf16_t* __restrict__ A, int lda,
    const bf16_t* __restrict__ W, int ldw,
    const float* __restrict__ bias,
    float* __restrict__ Yf, bf16_t* __restrict__ Ybf, int ldy,
    int M, int N, int K) {
#if USE_ASYNC_LDS
  __shared__ bf16_t sA[2 * TILE_ELEMS];
  __shared__ bf16_t sB[2 * TILE_ELEMS];
#else
  __shared__ bf16_t sA[TILE_ELEMS];
  __shared__ bf16_t sB[TILE_ELEMS];
#endif

  const int tid   = threadIdx.x;
  const int lane  = tid & 31;
  const int wid   = tid >> 5;
  const int waveM = wid >> 2;   // 0..1  -> 64-row slab
  const int waveN = wid & 3;    // 0..3  -> 32-col slab
  const int mBlk  = blockIdx.y * BM;
  const int nBlk  = blockIdx.x * BN;

  v8f acc[4][2];
#pragma unroll
  for (int i = 0; i < 4; ++i)
#pragma unroll
    for (int j = 0; j < 2; ++j) acc[i][j] = (v8f)(0.0f);

#if USE_ASYNC_LDS
  // Prologue: fill buffer 0, then ping-pong: issue async fill of the other
  // buffer, compute on the current one, wait + barrier.
  stage_async(A, lda, mBlk, W, ldw, nBlk, sA, sB, 0, tid);
  async_wait0();
  __syncthreads();
  for (int k0 = 0; k0 < K; k0 += BK) {
    const int cur = (k0 / BK) & 1;
    const int nxt = cur ^ 1;
    const bool more = (k0 + BK) < K;
    if (more)
      stage_async(A, lda, mBlk, W, ldw, nBlk,
                  sA + nxt * TILE_ELEMS, sB + nxt * TILE_ELEMS, k0 + BK, tid);
    do_wmma(sA + cur * TILE_ELEMS, sB + cur * TILE_ELEMS, acc, lane, waveM, waveN);
    if (more) async_wait0();
    __syncthreads();
  }
#else
  // Fallback: register-staged, software-pipelined (load next tile's 8 x b128
  // into dedicated VGPRs while WMMAs run on the current tile).
  uint4 ra[4], rb[4];
  {
#pragma unroll
    for (int i = 0; i < 4; ++i) {
      int idx = tid + i * 256;
      int row = idx >> 3, ch = (idx & 7) << 3;
      ra[i] = *(const uint4*)(A + (size_t)(mBlk + row) * lda + ch);
      rb[i] = *(const uint4*)(W + (size_t)(nBlk + row) * ldw + ch);
    }
  }
  for (int k0 = 0; k0 < K; k0 += BK) {
    const bool more = (k0 + BK) < K;
#pragma unroll
    for (int i = 0; i < 4; ++i) {
      int idx = tid + i * 256;
      int row = idx >> 3, ch = (idx & 7) << 3;
      *(uint4*)(sA + row * LDT + ch) = ra[i];
      *(uint4*)(sB + row * LDT + ch) = rb[i];
    }
    __syncthreads();
    if (more) {
#pragma unroll
      for (int i = 0; i < 4; ++i) {
        int idx = tid + i * 256;
        int row = idx >> 3, ch = (idx & 7) << 3;
        ra[i] = *(const uint4*)(A + (size_t)(mBlk + row) * lda + k0 + BK + ch);
        rb[i] = *(const uint4*)(W + (size_t)(nBlk + row) * ldw + k0 + BK + ch);
      }
    }
    do_wmma(sA, sB, acc, lane, waveM, waveN);
    __syncthreads();
  }
#endif

  // Epilogue: C/D 16x16 f32 layout: lane -> col L%16, VGPR r -> row r+8*(L/16)
  const int nC    = lane & 15;
  const int mHalf = (lane >> 4) << 3;
#pragma unroll
  for (int i = 0; i < 4; ++i) {
#pragma unroll
    for (int j = 0; j < 2; ++j) {
      int n = nBlk + waveN * 32 + j * 16 + nC;
      float bv = bias ? bias[n] : 0.0f;
#pragma unroll
      for (int r = 0; r < 8; ++r) {
        int m = mBlk + waveM * 64 + i * 16 + mHalf + r;
        float y = acc[i][j][r] + bv;
        if (Yf)  Yf[(size_t)m * ldy + n]  = y;
        if (Ybf) Ybf[(size_t)m * ldy + n] = f2bf(y);
      }
    }
  }
}

// ---------------------------------------------------------------------------
// S=2 attention: one wave32 per (b, h). D=128 -> 4 dims per lane.
// ---------------------------------------------------------------------------
__device__ __forceinline__ float wave_sum(float x) {
#pragma unroll
  for (int o = 16; o > 0; o >>= 1) x += __shfl_xor(x, o, 32);
  return x;
}

__global__ __launch_bounds__(256) void attn_s2(
    const bf16_t* __restrict__ Q, int qld,
    const bf16_t* __restrict__ Kp, int kld,
    const bf16_t* __restrict__ Vp, int vld,
    bf16_t* __restrict__ O, int outld, int BH) {
  int gw   = (int)((blockIdx.x * 256u + threadIdx.x) >> 5);
  int lane = threadIdx.x & 31;
  if (gw >= BH) return;
  int b = gw >> 4, h = gw & 15;
  int col = h * HD + lane * 4;
  size_t r0 = (size_t)(2 * b), r1 = r0 + 1;

  float q0[4], q1[4], k0[4], k1[4], v0[4], v1[4];
  {
    const bf16_t* p;
    p = Q  + r0 * qld + col;  for (int i = 0; i < 4; ++i) q0[i] = bf2f(p[i]);
    p = Q  + r1 * qld + col;  for (int i = 0; i < 4; ++i) q1[i] = bf2f(p[i]);
    p = Kp + r0 * kld + col;  for (int i = 0; i < 4; ++i) k0[i] = bf2f(p[i]);
    p = Kp + r1 * kld + col;  for (int i = 0; i < 4; ++i) k1[i] = bf2f(p[i]);
    p = Vp + r0 * vld + col;  for (int i = 0; i < 4; ++i) v0[i] = bf2f(p[i]);
    p = Vp + r1 * vld + col;  for (int i = 0; i < 4; ++i) v1[i] = bf2f(p[i]);
  }
  float d00 = 0, d01 = 0, d10 = 0, d11 = 0;
#pragma unroll
  for (int i = 0; i < 4; ++i) {
    d00 += q0[i] * k0[i];  d01 += q0[i] * k1[i];
    d10 += q1[i] * k0[i];  d11 += q1[i] * k1[i];
  }
  const float sc = 0.088388347648318447f;  // 1/sqrt(128)
  d00 = wave_sum(d00) * sc;  d01 = wave_sum(d01) * sc;
  d10 = wave_sum(d10) * sc;  d11 = wave_sum(d11) * sc;

  float m0 = fmaxf(d00, d01), m1 = fmaxf(d10, d11);
  float e00 = __expf(d00 - m0), e01 = __expf(d01 - m0);
  float e10 = __expf(d10 - m1), e11 = __expf(d11 - m1);
  float i0 = 1.0f / (e00 + e01), i1 = 1.0f / (e10 + e11);
  float a00 = e00 * i0, a01 = e01 * i0;
  float a10 = e10 * i1, a11 = e11 * i1;

  bf16_t* o0 = O + r0 * outld + col;
  bf16_t* o1 = O + r1 * outld + col;
#pragma unroll
  for (int i = 0; i < 4; ++i) {
    o0[i] = f2bf(a00 * v0[i] + a01 * v1[i]);
    o1[i] = f2bf(a10 * v0[i] + a11 * v1[i]);
  }
}

// ---------------------------------------------------------------------------
// LayerNorm over C=2048; one 256-thread block per row; dual f32/bf16 output.
// ---------------------------------------------------------------------------
__global__ __launch_bounds__(256) void ln_kernel(
    const float* __restrict__ x, const float* __restrict__ g,
    const float* __restrict__ b, float* __restrict__ yf,
    bf16_t* __restrict__ ybf) {
  __shared__ float red[16];
  __shared__ float stats[2];
  const int row = blockIdx.x;
  const float* xr = x + (size_t)row * CDIM;
  float vals[8];
  float s = 0.0f, s2 = 0.0f;
#pragma unroll
  for (int i = 0; i < 8; ++i) {
    float v = xr[threadIdx.x + i * 256];
    vals[i] = v; s += v; s2 += v * v;
  }
#pragma unroll
  for (int o = 16; o > 0; o >>= 1) { s += __shfl_xor(s, o, 32); s2 += __shfl_xor(s2, o, 32); }
  int wid = threadIdx.x >> 5, lane = threadIdx.x & 31;
  if (lane == 0) { red[wid] = s; red[8 + wid] = s2; }
  __syncthreads();
  if (threadIdx.x == 0) {
    float ts = 0, ts2 = 0;
#pragma unroll
    for (int i = 0; i < 8; ++i) { ts += red[i]; ts2 += red[8 + i]; }
    float mu = ts * (1.0f / CDIM);
    float var = ts2 * (1.0f / CDIM) - mu * mu;
    stats[0] = mu; stats[1] = rsqrtf(var + 1e-5f);
  }
  __syncthreads();
  float mu = stats[0], rs = stats[1];
#pragma unroll
  for (int i = 0; i < 8; ++i) {
    int c = threadIdx.x + i * 256;
    float y = (vals[i] - mu) * rs * g[c] + b[c];
    yf[(size_t)row * CDIM + c]  = y;
    ybf[(size_t)row * CDIM + c] = f2bf(y);
  }
}

// ---------------------------------------------------------------------------
// fused = mean_s( ln + cross )  ->  bf16 for final GEMM
// ---------------------------------------------------------------------------
__global__ __launch_bounds__(256) void fuse_mean(
    const float* __restrict__ ln, const float* __restrict__ cr,
    bf16_t* __restrict__ out, size_t n) {
  size_t i = (size_t)blockIdx.x * blockDim.x + threadIdx.x;
  size_t stride = (size_t)gridDim.x * blockDim.x;
  for (; i < n; i += stride) {
    size_t bi = i / CDIM, c = i % CDIM;
    size_t p0 = (2 * bi) * (size_t)CDIM + c, p1 = p0 + CDIM;
    out[i] = f2bf(0.5f * ((ln[p0] + cr[p0]) + (ln[p1] + cr[p1])));
  }
}

// ---------------------------------------------------------------------------
// Host-side orchestration
// ---------------------------------------------------------------------------
static inline size_t alup(size_t x) { return (x + 255u) & ~(size_t)255u; }

extern "C" void kernel_launch(void* const* d_in, const int* in_sizes, int n_in,
                              void* d_out, int out_size, void* d_ws, size_t ws_size,
                              hipStream_t stream) {
  (void)in_sizes; (void)n_in; (void)out_size; (void)ws_size;

  const float* text   = (const float*)d_in[0];
  const float* graph  = (const float*)d_in[1];
  const float* Wt     = (const float*)d_in[2];
  const float* bt     = (const float*)d_in[3];
  const float* Wg     = (const float*)d_in[4];
  const float* bg     = (const float*)d_in[5];
  const float* Wqkv_s = (const float*)d_in[6];
  const float* bqkv_s = (const float*)d_in[7];
  const float* Wo_s   = (const float*)d_in[8];
  const float* bo_s   = (const float*)d_in[9];
  const float* Wqkv_c = (const float*)d_in[10];
  const float* bqkv_c = (const float*)d_in[11];
  const float* Wo_c   = (const float*)d_in[12];
  const float* bo_c   = (const float*)d_in[13];
  const float* ln_w   = (const float*)d_in[14];
  const float* ln_b   = (const float*)d_in[15];
  const float* Wout   = (const float*)d_in[16];
  const float* bout   = (const float*)d_in[17];

  // ---- workspace layout (bytes), arenas reused across phases ----
  char* ws = (char*)d_ws;
  size_t off = 0;
  auto take = [&](size_t bytes) { char* p = ws + off; off += alup(bytes); return p; };

  bf16_t* wt_bf    = (bf16_t*)take((size_t)CDIM * TDIM * 2);
  bf16_t* wg_bf    = (bf16_t*)take((size_t)CDIM * GDIM * 2);
  bf16_t* wqkvs_bf = (bf16_t*)take((size_t)3 * CDIM * CDIM * 2);
  bf16_t* wos_bf   = (bf16_t*)take((size_t)CDIM * CDIM * 2);
  bf16_t* wqkvc_bf = (bf16_t*)take((size_t)3 * CDIM * CDIM * 2);
  bf16_t* woc_bf   = (bf16_t*)take((size_t)CDIM * CDIM * 2);
  bf16_t* wout_bf  = (bf16_t*)take((size_t)ODIM * CDIM * 2);

  bf16_t* stacked_bf = (bf16_t*)take((size_t)ROWS * CDIM * 2);
  float*  ln_f32     = (float*) take((size_t)ROWS * CDIM * 4);
  bf16_t* ln_bf      = (bf16_t*)take((size_t)ROWS * CDIM * 2);

  // Arena 1: xt_bf + xg_bf  (phase 1)  ->  fused_bf (phase 4)
  char* ar1 = take((size_t)BDIM * TDIM * 2 + (size_t)BDIM * GDIM * 2);
  bf16_t* xt_bf    = (bf16_t*)ar1;
  bf16_t* xg_bf    = (bf16_t*)(ar1 + (size_t)BDIM * TDIM * 2);
  bf16_t* fused_bf = (bf16_t*)ar1;

  // Arena 2: qkv_s (phase 2) -> qc + kvc (phase 3)
  char* ar2 = take((size_t)ROWS * 3 * CDIM * 2);
  bf16_t* qkvs_bf = (bf16_t*)ar2;
  bf16_t* qc_bf   = (bf16_t*)ar2;
  bf16_t* kvc_bf  = (bf16_t*)(ar2 + (size_t)ROWS * CDIM * 2);

  // Arena 3: attn_self -> attn_cross
  char* ar3 = take((size_t)ROWS * CDIM * 2);
  bf16_t* attnS_bf = (bf16_t*)ar3;
  bf16_t* attnC_bf = (bf16_t*)ar3;

  // Arena 4: selfproj f32 -> cross f32
  char* ar4 = take((size_t)ROWS * CDIM * 4);
  float* self_f32  = (float*)ar4;
  float* cross_f32 = (float*)ar4;

  // ---- helpers ----
  auto cvt = [&](const float* in, bf16_t* out, size_t n) {
    int blocks = (int)((n + 255) / 256);
    if (blocks > 4096) blocks = 4096;
    cvt_f32_bf16<<<blocks, 256, 0, stream>>>(in, out, n);
  };
  auto gemm = [&](const bf16_t* A, int lda, const bf16_t* W, int ldw,
                  const float* bias, float* Yf, bf16_t* Ybf, int ldy,
                  int M, int N, int K) {
    dim3 g(N / BN, M / BM);
    gemm_bf16_wmma<<<g, 256, 0, stream>>>(A, lda, W, ldw, bias, Yf, Ybf, ldy, M, N, K);
  };

  // ---- 0) precision downcasts ----
  cvt(Wt,     wt_bf,    (size_t)CDIM * TDIM);
  cvt(Wg,     wg_bf,    (size_t)CDIM * GDIM);
  cvt(Wqkv_s, wqkvs_bf, (size_t)3 * CDIM * CDIM);
  cvt(Wo_s,   wos_bf,   (size_t)CDIM * CDIM);
  cvt(Wqkv_c, wqkvc_bf, (size_t)3 * CDIM * CDIM);
  cvt(Wo_c,   woc_bf,   (size_t)CDIM * CDIM);
  cvt(Wout,   wout_bf,  (size_t)ODIM * CDIM);
  cvt(text,   xt_bf,    (size_t)BDIM * TDIM);
  cvt(graph,  xg_bf,    (size_t)BDIM * GDIM);

  // ---- 1) input projections, interleaved into stacked [B,2,C] ----
  gemm(xt_bf, TDIM, wt_bf, TDIM, bt, nullptr, stacked_bf,        2 * CDIM, BDIM, CDIM, TDIM);
  gemm(xg_bf, GDIM, wg_bf, GDIM, bg, nullptr, stacked_bf + CDIM, 2 * CDIM, BDIM, CDIM, GDIM);

  // ---- 2) self-attention ----
  gemm(stacked_bf, CDIM, wqkvs_bf, CDIM, bqkv_s, nullptr, qkvs_bf, 3 * CDIM,
       ROWS, 3 * CDIM, CDIM);
  {
    int BH = BDIM * HEADS;
    attn_s2<<<(BH * 32 + 255) / 256, 256, 0, stream>>>(
        qkvs_bf, 3 * CDIM, qkvs_bf + CDIM, 3 * CDIM, qkvs_bf + 2 * CDIM, 3 * CDIM,
        attnS_bf, CDIM, BH);
  }
  gemm(attnS_bf, CDIM, wos_bf, CDIM, bo_s, self_f32, nullptr, CDIM,
       ROWS, CDIM, CDIM);

  // LayerNorm -> ln_f32 (for residual) + ln_bf (for Q_c GEMM)
  ln_kernel<<<ROWS, 256, 0, stream>>>(self_f32, ln_w, ln_b, ln_f32, ln_bf);

  // ---- 3) cross-attention (Q from ln, K/V from stacked) ----
  gemm(ln_bf, CDIM, wqkvc_bf, CDIM, bqkv_c, nullptr, qc_bf, CDIM,
       ROWS, CDIM, CDIM);
  gemm(stacked_bf, CDIM, wqkvc_bf + (size_t)CDIM * CDIM, CDIM, bqkv_c + CDIM,
       nullptr, kvc_bf, 2 * CDIM, ROWS, 2 * CDIM, CDIM);
  {
    int BH = BDIM * HEADS;
    attn_s2<<<(BH * 32 + 255) / 256, 256, 0, stream>>>(
        qc_bf, CDIM, kvc_bf, 2 * CDIM, kvc_bf + CDIM, 2 * CDIM,
        attnC_bf, CDIM, BH);
  }
  gemm(attnC_bf, CDIM, woc_bf, CDIM, bo_c, cross_f32, nullptr, CDIM,
       ROWS, CDIM, CDIM);

  // ---- 4) residual + mean over S, then output head ----
  {
    size_t n = (size_t)BDIM * CDIM;
    fuse_mean<<<4096, 256, 0, stream>>>(ln_f32, cross_f32, fused_bf, n);
  }
  gemm(fused_bf, CDIM, wout_bf, CDIM, bout, (float*)d_out, nullptr, ODIM,
       BDIM, ODIM, CDIM);
}
